// Evolution_13889924235705
// MI455X (gfx1250) — matrix operand order repaired and checked by reading, not compiled, not run on hardware
//
#include <hip/hip_runtime.h>
#include <hip/hip_bf16.h>

typedef __attribute__((ext_vector_type(16))) _Float16 v16h;
typedef __attribute__((ext_vector_type(8)))  _Float16 h8;
typedef __attribute__((ext_vector_type(8)))  float    v8f;

#define NPOLY 512
#define NODES 100
#define NPOS  (NPOLY * NODES)   // 51200
#define CFEAT 36
#define STATE 128
#define HH 512
#define WW 512
#define BN_EPS 1e-5f
#define CLIP_DIS 16.0f

// Build a v16h fragment from two contiguous 8-half (16B) runs.
__device__ inline v16h mk16(h8 lo, h8 hi) {
  v16h f;
#pragma unroll
  for (int i = 0; i < 8; ++i) { f[i] = lo[i]; f[i + 8] = hi[i]; }
  return f;
}

// ---------------- weight conversion: plain f32 -> f16 ----------------
__global__ void cvt_f16_kernel(const float* __restrict__ in, _Float16* __restrict__ out, int n) {
  int i = blockIdx.x * blockDim.x + threadIdx.x;
  if (i < n) out[i] = (_Float16)in[i];
}

// res weights (7,128,128,9) [l][co][ci][t] -> f16 [l][co][t*128+ci]  (K reordered t-major)
__global__ void cvt_res_w_kernel(const float* __restrict__ in, _Float16* __restrict__ out) {
  int i = blockIdx.x * blockDim.x + threadIdx.x;
  if (i >= 7 * 128 * 1152) return;
  int lco = i / 1152;            // layer*128 + co
  int k   = i - lco * 1152;      // t*128 + ci
  int t = k >> 7, ci = k & 127;
  out[i] = (_Float16)in[(size_t)lco * 1152 + ci * 9 + t];
}

// ---------------- bilinear gather: feats[pos][36] ----------------
__global__ void gather_kernel(const float* __restrict__ cnn, const float* __restrict__ py,
                              const int* __restrict__ inds, float* __restrict__ feats) {
  int pos = blockIdx.x * blockDim.x + threadIdx.x;
  if (pos >= NPOS) return;
  int poly = pos / NODES;
  float xx = py[pos * 2 + 0] - 0.5f;
  float yy = py[pos * 2 + 1] - 0.5f;
  float xf = floorf(xx), yf = floorf(yy);
  int x0 = (int)xf, y0 = (int)yf;
  float wx1 = xx - xf, wy1 = yy - yf;
  float wx0 = 1.f - wx1, wy0 = 1.f - wy1;
  int b = inds[poly];
  const float* base = cnn + (size_t)b * CFEAT * HH * WW;
  int   xs[2]  = {x0, x0 + 1};
  int   ys[2]  = {y0, y0 + 1};
  float wxs[2] = {wx0, wx1};
  float wys[2] = {wy0, wy1};
  for (int c = 0; c < CFEAT; ++c) {
    const float* cb = base + (size_t)c * HH * WW;
    float s = 0.f;
#pragma unroll
    for (int ty = 0; ty < 2; ++ty)
#pragma unroll
      for (int tx = 0; tx < 2; ++tx) {
        int xi = xs[tx], yi = ys[ty];
        bool valid = (xi >= 0) && (xi < WW) && (yi >= 0) && (yi < HH);
        int xc = min(max(xi, 0), WW - 1), yc = min(max(yi, 0), HH - 1);
        float v = cb[yc * WW + xc];
        s += valid ? v * (wys[ty] * wxs[tx]) : 0.f;
      }
    feats[pos * CFEAT + c] = s;
  }
}

// ---------------- head circular conv (K=324, scalar) -> relu -> yb[pos][128] ----------------
__global__ void head_conv_kernel(const float* __restrict__ feats, const float* __restrict__ W,
                                 const float* __restrict__ bias, float* __restrict__ yb) {
  int idx = blockIdx.x * blockDim.x + threadIdx.x;
  if (idx >= NPOS * STATE) return;
  int pos = idx >> 7;
  int co  = idx & 127;
  int poly = pos / NODES, l = pos % NODES;
  float acc = bias[co];
  for (int t = 0; t < 9; ++t) {
    int sl = (l + t - 4 + NODES) % NODES;
    const float* fp = feats + (size_t)(poly * NODES + sl) * CFEAT;
    const float* wp = W + (size_t)co * CFEAT * 9 + t;
    for (int ci = 0; ci < CFEAT; ++ci) acc = fmaf(fp[ci], wp[ci * 9], acc);
  }
  yb[idx] = fmaxf(acc, 0.f);
}

// ---------------- BN stats over 51200 samples per channel ----------------
__global__ void bn_stats_kernel(const float* __restrict__ yb, float* __restrict__ mu,
                                float* __restrict__ rs) {
  int c = blockIdx.x;
  __shared__ float ss[256], sq[256];
  float s = 0.f, q = 0.f;
  for (int pos = threadIdx.x; pos < NPOS; pos += 256) {
    float v = yb[(size_t)pos * STATE + c];
    s += v; q += v * v;
  }
  ss[threadIdx.x] = s; sq[threadIdx.x] = q;
  __syncthreads();
  for (int st = 128; st > 0; st >>= 1) {
    if (threadIdx.x < st) { ss[threadIdx.x] += ss[threadIdx.x + st]; sq[threadIdx.x] += sq[threadIdx.x + st]; }
    __syncthreads();
  }
  if (threadIdx.x == 0) {
    float m = ss[0] / (float)NPOS;
    float var = sq[0] / (float)NPOS - m * m;
    mu[c] = m;
    rs[c] = rsqrtf(var + BN_EPS);
  }
}

// ---------------- BN apply (+residual), write f32 x and f16 state slot ----------------
__global__ void bn_apply_kernel(const float* __restrict__ yb, const float* __restrict__ mu,
                                const float* __restrict__ rs, const float* __restrict__ gam,
                                const float* __restrict__ bet, float* __restrict__ x,
                                _Float16* __restrict__ state, int slot, int residual) {
  int idx = blockIdx.x * blockDim.x + threadIdx.x;
  if (idx >= NPOS * STATE) return;
  int c = idx & 127;
  int pos = idx >> 7;
  float v = (yb[idx] - mu[c]) * rs[c] * gam[c] + bet[c];
  if (residual) v += x[idx];
  x[idx] = v;
  state[(size_t)pos * 1024 + slot * STATE + c] = (_Float16)v;
}

// ---------------- residual circular conv: LDS-free WMMA GEMM ----------------
// K = 1152 reordered as t*128+ci; one wave = one (M-tile, co-pair) task; frags loaded
// directly from global with 16B vector loads. blockDim=256 (8 waves).
__global__ void res_conv_kernel(const _Float16* __restrict__ state, int slot,
                                const _Float16* __restrict__ Wt,  // [128][1152] t-major f16
                                const float* __restrict__ bias, int dil,
                                float* __restrict__ yb) {
  int lane = threadIdx.x & 31;
  int wv = (blockIdx.x * blockDim.x + threadIdx.x) >> 5;  // 0..12799
  int mt = wv >> 2;         // M-tile 0..3199
  int cp = wv & 3;          // co-pair 0..3
  int m0 = mt * 16;
  int r = lane & 15;
  int pos = m0 + r;
  int p = pos / NODES;
  int l = pos - p * NODES;
  int kbA = (lane & 16) ? 8 : 0;
  int n = lane & 15;
  int kbB = (lane & 16) ? 8 : 0;
  int co0 = cp * 32 + n;
  int co1 = cp * 32 + 16 + n;
  const _Float16* b0row = Wt + (size_t)co0 * 1152;
  const _Float16* b1row = Wt + (size_t)co1 * 1152;
  const _Float16* srow = state + (size_t)slot * STATE;  // + pos*1024 later
  v8f acc0, acc1;
#pragma unroll
  for (int i = 0; i < 8; ++i) { acc0[i] = 0.f; acc1[i] = 0.f; }
  for (int ks = 0; ks < 36; ++ks) {
    int t = ks >> 2;
    int ci0 = (ks & 3) * 32;
    int kk0 = ks * 32;
    int sl = l + (t - 4) * dil;
    sl = (sl < 0) ? sl + NODES : sl;
    sl = (sl >= NODES) ? sl - NODES : sl;
    const _Float16* ap = srow + (size_t)(p * NODES + sl) * 1024 + ci0 + kbA;
    v16h a = mk16(*(const h8*)ap, *(const h8*)(ap + 16));
    const _Float16* bp0 = b0row + kk0 + kbB;
    v16h b0 = mk16(*(const h8*)bp0, *(const h8*)(bp0 + 16));
    acc0 = __builtin_amdgcn_wmma_f32_16x16x32_f16(false, a, false, b0, (short)0, acc0, false, false);
    const _Float16* bp1 = b1row + kk0 + kbB;
    v16h b1 = mk16(*(const h8*)bp1, *(const h8*)(bp1 + 16));
    acc1 = __builtin_amdgcn_wmma_f32_16x16x32_f16(false, a, false, b1, (short)0, acc1, false, false);
  }
  int mo = (lane & 16) ? 8 : 0;
#pragma unroll
  for (int rr = 0; rr < 8; ++rr) {
    int opos = m0 + rr + mo;
    yb[(size_t)opos * STATE + co0] = fmaxf(acc0[rr] + bias[co0], 0.f);
    yb[(size_t)opos * STATE + co1] = fmaxf(acc1[rr] + bias[co1], 0.f);
  }
}

// ---------------- dense WMMA GEMM (LDS-free): out[pos][N] = A[pos][K]*Wt[N][K]^T + b ----
// optional f16 G prefix (p0: K<256 reads gh[poly][k]); TPW co-tiles per wave.
template <int TPW>
__global__ void dense_gemm_kernel(const _Float16* __restrict__ A, int lda,
                                  const _Float16* __restrict__ G,
                                  const _Float16* __restrict__ Wt, const float* __restrict__ bias,
                                  int N, int K, int relu,
                                  float* __restrict__ outF, _Float16* __restrict__ outH) {
  int lane = threadIdx.x & 31;
  int wv = (blockIdx.x * blockDim.x + threadIdx.x) >> 5;
  int tilesPerM = (N / 16) / TPW;
  int mt = wv / tilesPerM;
  int tg = wv - mt * tilesPerM;
  int m0 = mt * 16;
  int r = lane & 15;
  int pos = m0 + r;
  int poly = pos / NODES;
  int kbA = (lane & 16) ? 8 : 0;
  int n = lane & 15;
  int kbB = (lane & 16) ? 8 : 0;
  int koff = (G != nullptr) ? 256 : 0;
  const _Float16* arow = A + (size_t)pos * lda;
  const _Float16* grow = (G != nullptr) ? (G + (size_t)poly * 256) : A;
  v8f acc[TPW];
#pragma unroll
  for (int t = 0; t < TPW; ++t)
#pragma unroll
    for (int i = 0; i < 8; ++i) acc[t][i] = 0.f;
  for (int ks = 0; ks * 32 < K; ++ks) {
    int kk0 = ks * 32;
    const _Float16* ap = (kk0 < koff) ? (grow + kk0 + kbA)
                                      : (arow + (kk0 - koff) + kbA);
    v16h a = mk16(*(const h8*)ap, *(const h8*)(ap + 16));
#pragma unroll
    for (int t = 0; t < TPW; ++t) {
      int co = (tg * TPW + t) * 16 + n;
      const _Float16* bp = Wt + (size_t)co * K + kk0 + kbB;
      v16h b = mk16(*(const h8*)bp, *(const h8*)(bp + 16));
      acc[t] = __builtin_amdgcn_wmma_f32_16x16x32_f16(false, a, false, b, (short)0, acc[t], false, false);
    }
  }
  int mo = (lane & 16) ? 8 : 0;
#pragma unroll
  for (int t = 0; t < TPW; ++t) {
    int co = (tg * TPW + t) * 16 + n;
    float bsv = bias[co];
#pragma unroll
    for (int rr = 0; rr < 8; ++rr) {
      int opos = m0 + rr + mo;
      float v = acc[t][rr] + bsv;
      if (relu) v = fmaxf(v, 0.f);
      if (outF) outF[(size_t)opos * N + co] = v;
      else      outH[(size_t)opos * N + co] = (_Float16)v;
    }
  }
}

// ---------------- per-poly max over length (fusion -> gh f16) ----------------
__global__ void max_reduce_kernel(const float* __restrict__ fus, _Float16* __restrict__ gh) {
  int t = blockIdx.x * blockDim.x + threadIdx.x;
  if (t >= NPOLY * 256) return;
  int poly = t >> 8, co = t & 255;
  float m = -3.4e38f;
  const float* p = fus + (size_t)poly * NODES * 256 + co;
  for (int l = 0; l < NODES; ++l) m = fmaxf(m, p[l * 256]);
  gh[t] = (_Float16)m;
}

// ---------------- final 64->2 projection + coordinate update ----------------
__global__ void p2_update_kernel(const _Float16* __restrict__ h2, const float* __restrict__ w2,
                                 const float* __restrict__ b2, float* __restrict__ py,
                                 float* __restrict__ pred) {
  int pos = blockIdx.x * blockDim.x + threadIdx.x;
  if (pos >= NPOS) return;
  float o0 = b2[0], o1 = b2[1];
  const _Float16* h = h2 + (size_t)pos * 64;
  for (int j = 0; j < 64; ++j) {
    float hv = (float)h[j];
    o0 = fmaf(hv, w2[j], o0);
    o1 = fmaf(hv, w2[64 + j], o1);
  }
  o0 = fminf(fmaxf(o0, -CLIP_DIS), CLIP_DIS);
  o1 = fminf(fmaxf(o1, -CLIP_DIS), CLIP_DIS);
  float nx = fminf(fmaxf(py[pos * 2 + 0] + o0, 1.f), (float)(WW - 2));
  float ny = fminf(fmaxf(py[pos * 2 + 1] + o1, 1.f), (float)(WW - 2));
  py[pos * 2 + 0] = nx; py[pos * 2 + 1] = ny;
  pred[pos * 2 + 0] = nx; pred[pos * 2 + 1] = ny;
}

extern "C" void kernel_launch(void* const* d_in, const int* in_sizes, int n_in,
                              void* d_out, int out_size, void* d_ws, size_t ws_size,
                              hipStream_t stream) {
  (void)in_sizes; (void)n_in; (void)out_size; (void)ws_size;
  const float* cnn        = (const float*)d_in[0];
  const float* init_polys = (const float*)d_in[1];
  const int*   inds       = (const int*)d_in[2];

  char* ws = (char*)d_ws;
  size_t o = 0;
  auto take = [&](size_t bytes) -> void* {
    o = (o + 255) & ~(size_t)255;
    void* p = ws + o;
    o += bytes;
    return p;
  };
  float*    py    = (float*)take((size_t)NPOS * 2 * 4);
  float*    feats = (float*)take((size_t)NPOS * CFEAT * 4);
  float*    x     = (float*)take((size_t)NPOS * STATE * 4);
  float*    yb    = (float*)take((size_t)NPOS * STATE * 4);
  _Float16* state = (_Float16*)take((size_t)NPOS * 1024 * 2);
  float*    mu    = (float*)take(STATE * 4);
  float*    rs    = (float*)take(STATE * 4);
  float*    fus   = (float*)take((size_t)NPOS * 256 * 4);
  _Float16* gh    = (_Float16*)take((size_t)NPOLY * 256 * 2);
  _Float16* h1    = (_Float16*)take((size_t)NPOS * 256 * 2);
  _Float16* h2    = (_Float16*)take((size_t)NPOS * 64 * 2);
  _Float16* wt_res = (_Float16*)take((size_t)7 * 128 * 1152 * 2);
  _Float16* wt_fus = (_Float16*)take((size_t)256 * 1024 * 2);
  _Float16* wt_p0  = (_Float16*)take((size_t)256 * 1280 * 2);
  _Float16* wt_p1  = (_Float16*)take((size_t)64 * 256 * 2);

  hipMemcpyAsync(py, init_polys, (size_t)NPOS * 2 * 4, hipMemcpyDeviceToDevice, stream);

  static const int DIL[7] = {1, 1, 1, 2, 2, 4, 4};

  for (int it = 0; it < 3; ++it) {
    const float* pp[16];
    for (int j = 0; j < 16; ++j) pp[j] = (const float*)d_in[3 + it * 16 + j];
    const float* head_w = pp[0];  const float* head_b = pp[1];
    const float* head_g = pp[2];  const float* head_be = pp[3];
    const float* res_w  = pp[4];  const float* res_b  = pp[5];
    const float* res_g  = pp[6];  const float* res_be = pp[7];
    const float* fus_w  = pp[8];  const float* fus_b  = pp[9];
    const float* p0_w   = pp[10]; const float* p0_b   = pp[11];
    const float* p1_w   = pp[12]; const float* p1_b   = pp[13];
    const float* p2_w   = pp[14]; const float* p2_b   = pp[15];

    // per-iteration weight conversion to f16 (res: K reordered t-major)
    cvt_res_w_kernel<<<(7 * 128 * 1152 + 255) / 256, 256, 0, stream>>>(res_w, wt_res);
    cvt_f16_kernel<<<(256 * 1024 + 255) / 256, 256, 0, stream>>>(fus_w, wt_fus, 256 * 1024);
    cvt_f16_kernel<<<(256 * 1280 + 255) / 256, 256, 0, stream>>>(p0_w, wt_p0, 256 * 1280);
    cvt_f16_kernel<<<(64 * 256 + 255) / 256, 256, 0, stream>>>(p1_w, wt_p1, 64 * 256);

    gather_kernel<<<NPOS / 256, 256, 0, stream>>>(cnn, py, inds, feats);
    head_conv_kernel<<<NPOS * STATE / 256, 256, 0, stream>>>(feats, head_w, head_b, yb);
    bn_stats_kernel<<<STATE, 256, 0, stream>>>(yb, mu, rs);
    bn_apply_kernel<<<NPOS * STATE / 256, 256, 0, stream>>>(yb, mu, rs, head_g, head_be, x, state, 0, 0);

    for (int i = 0; i < 7; ++i) {
      res_conv_kernel<<<12800 / 8, 256, 0, stream>>>(state, i,
          wt_res + (size_t)i * 128 * 1152, res_b + i * STATE, DIL[i], yb);
      bn_stats_kernel<<<STATE, 256, 0, stream>>>(yb, mu, rs);
      bn_apply_kernel<<<NPOS * STATE / 256, 256, 0, stream>>>(yb, mu, rs,
          res_g + i * STATE, res_be + i * STATE, x, state, i + 1, 1);
    }

    // fusion (K=1024, N=256) -> f32, then per-poly max -> gh (f16)
    dense_gemm_kernel<2><<<3200, 256, 0, stream>>>(state, 1024, (const _Float16*)nullptr,
        wt_fus, fus_b, 256, 1024, 0, fus, (_Float16*)nullptr);
    max_reduce_kernel<<<NPOLY * 256 / 256, 256, 0, stream>>>(fus, gh);
    // p0: A = [gh | state], K=1280, N=256, relu -> h1 (f16)
    dense_gemm_kernel<2><<<3200, 256, 0, stream>>>(state, 1024, gh,
        wt_p0, p0_b, 256, 1280, 1, (float*)nullptr, h1);
    // p1: K=256, N=64, relu -> h2 (f16)
    dense_gemm_kernel<1><<<1600, 256, 0, stream>>>(h1, 256, (const _Float16*)nullptr,
        wt_p1, p1_b, 64, 256, 1, (float*)nullptr, h2);
    // p2 + clip + poly update + write preds[it]
    p2_update_kernel<<<NPOS / 256, 256, 0, stream>>>(h2, p2_w, p2_b, py,
        (float*)d_out + (size_t)it * NPOS * 2);
  }
}
